// TransformerLayer_51685636440708
// MI455X (gfx1250) — matrix operand order repaired and checked by reading, not compiled, not run on hardware
//
#include <hip/hip_runtime.h>
#include <cstdint>

typedef __attribute__((ext_vector_type(16))) __bf16 v16bf;
typedef __attribute__((ext_vector_type(8)))  __bf16 v8bf;
typedef __attribute__((ext_vector_type(8)))  float  v8f;

// ---------------- constants ----------------
#define CC     256
#define LSEQ   4096
#define NB     2
#define MTOK   8192          // NB*LSEQ
#define ASTRIDE 40           // 32 + 8 pad (halfwords)
#define BSTRIDE 264          // 256 + 8 pad (halfwords)

// ---------------- device helpers ----------------
__device__ __forceinline__ unsigned short f2bf(float f) {
  unsigned int u = __float_as_uint(f);
  u += 0x7fffu + ((u >> 16) & 1u);
  return (unsigned short)(u >> 16);
}

__device__ __forceinline__ v16bf catbf(v8bf lo, v8bf hi) {
  return __builtin_shufflevector(lo, hi, 0,1,2,3,4,5,6,7,8,9,10,11,12,13,14,15);
}

// A fragment: 16x32 bf16. lane&15 = row M, lane>>4 selects K-halves {0..7,16..23} vs {8..15,24..31}
__device__ __forceinline__ v16bf frag_a(const unsigned short* As, int mOff, int lane) {
  int r  = mOff + (lane & 15);
  int kh = (lane >> 4) << 3;
  const v8bf* p0 = (const v8bf*)(As + r * ASTRIDE + kh);
  const v8bf* p1 = (const v8bf*)(As + r * ASTRIDE + 16 + kh);
  return catbf(*p0, *p1);
}
// B fragment: 32x16 bf16. lane&15 = col N, lane>>4 selects K-halves {0..7,16..23} vs {8..15,24..31}
__device__ __forceinline__ v16bf frag_b(const unsigned short* Bs, int nOff, int lane) {
  int n  = nOff + (lane & 15);
  int kh = (lane >> 4) << 3;
  v16bf b;
#pragma unroll
  for (int s = 0; s < 8; s++) {
    b[s]     = __builtin_bit_cast(__bf16, Bs[(kh + s) * BSTRIDE + n]);
    b[s + 8] = __builtin_bit_cast(__bf16, Bs[(16 + kh + s) * BSTRIDE + n]);
  }
  return b;
}

__device__ __forceinline__ v8f wmma_bf16(v16bf a, v16bf b, v8f c) {
  return __builtin_amdgcn_wmma_f32_16x16x32_bf16(false, a, false, b, (short)0, c, false, false);
}

template<int ACT> __device__ __forceinline__ float activate(float x) {
  if      constexpr (ACT == 1) return fmaxf(x, 0.f);                                   // relu
  else if constexpr (ACT == 2) return 0.5f * x * (1.f + erff(x * 0.7071067811865475f));// exact gelu
  else if constexpr (ACT == 3) return 1.f / (1.f + expf(-x));                          // sigmoid
  else if constexpr (ACT == 4) return tanhf(x);                                        // tanh
  else return x;
}

__device__ __forceinline__ float wred_sum(float v) {
  for (int o = 16; o; o >>= 1) v += __shfl_xor(v, o, 32);
  return v;
}
__device__ __forceinline__ float wred_max(float v) {
  for (int o = 16; o; o >>= 1) v = fmaxf(v, __shfl_xor(v, o, 32));
  return v;
}

// ---------------- generic WMMA GEMM ----------------
// Block tile 64x256x32, 8 waves, each wave 32x64 (2x4 WMMA frags -> 8 wmma / K-step).
// C[M,N] = act( concatK(A0[M,K0], A1[M,K-K0]) @ B[K,N] + bias )
// batched over blockIdx.z with element strides sA (A0,A1), sB, sC.
template<int ACT>
__global__ __launch_bounds__(256)
void gemm_kernel(const float* __restrict__ A0, const float* __restrict__ A1, int K0,
                 const float* __restrict__ Bw, const float* __restrict__ bias,
                 float* __restrict__ Cout, int M, int N, int K,
                 long long sA, long long sB, long long sC) {
  __shared__ alignas(16) unsigned short As[64 * ASTRIDE];
  __shared__ alignas(16) unsigned short Bs[32 * BSTRIDE];
  const int tid = threadIdx.x;
  const int z   = blockIdx.z;
  A0 += (size_t)z * sA;  A1 += (size_t)z * sA;
  Bw += (size_t)z * sB;  Cout += (size_t)z * sC;
  const int mBlock = blockIdx.y * 64;
  const int nBlock = blockIdx.x * 256;
  const int K1  = K - K0;
  const int ar  = tid >> 2, ac = (tid & 3) * 8;        // A loader: row, 8 k's
  const int bkr = tid >> 3, bng = (tid & 7) * 32;      // B loader: k row, 32 n's
  const int wave = tid >> 5, lane = tid & 31;
  const int wm = (wave >> 2) * 32, wn = (wave & 3) * 64;

  v8f acc[2][4];
#pragma unroll
  for (int i = 0; i < 2; i++)
#pragma unroll
    for (int j = 0; j < 4; j++)
#pragma unroll
      for (int v = 0; v < 8; v++) acc[i][j][v] = 0.f;

  const int rowG = mBlock + ar;
  for (int kb = 0; kb < K; kb += 32) {
#pragma unroll
    for (int j = 0; j < 8; j++) {
      int kk = kb + ac + j;
      const float* ap; size_t idx;
      if (kk < K0) { ap = A0; idx = (size_t)rowG * K0 + kk; }
      else         { ap = A1; idx = (size_t)rowG * K1 + (kk - K0); }
      As[ar * ASTRIDE + ac + j] = f2bf(ap[idx]);
    }
#pragma unroll
    for (int j = 0; j < 32; j++)
      Bs[bkr * BSTRIDE + bng + j] = f2bf(Bw[(size_t)(kb + bkr) * N + nBlock + bng + j]);
    __syncthreads();

    v16bf a0 = frag_a(As, wm, lane), a1 = frag_a(As, wm + 16, lane);
    v16bf bf[4];
#pragma unroll
    for (int f = 0; f < 4; f++) bf[f] = frag_b(Bs, wn + f * 16, lane);
#pragma unroll
    for (int f = 0; f < 4; f++) {
      acc[0][f] = wmma_bf16(a0, bf[f], acc[0][f]);
      acc[1][f] = wmma_bf16(a1, bf[f], acc[1][f]);
    }
    __syncthreads();
  }

#pragma unroll
  for (int fm = 0; fm < 2; fm++)
#pragma unroll
    for (int fn = 0; fn < 4; fn++)
#pragma unroll
      for (int v = 0; v < 8; v++) {
        int mrow = mBlock + wm + fm * 16 + v + ((lane >> 4) << 3);
        int col  = nBlock + wn + fn * 16 + (lane & 15);
        float x = acc[fm][fn][v];
        if (bias) x += bias[col];
        Cout[(size_t)mrow * N + col] = activate<ACT>(x);
      }
}

// ---------------- implicit-GEMM conv (NHWC, H=W=64, Cout=256) ----------------
// out[b,h,w,o] = act( sum_{ky,kx,ci} w[o,ci,ky,kx] * x[b, h+ky-ph, w+kx-pw, ci] + bias[o] )
// x channels: ci<256 -> x0, else x1 (concat).  K = KH*KW*(256+C1).
template<int ACT>
__global__ __launch_bounds__(256)
void conv_kernel(const float* __restrict__ x0, const float* __restrict__ x1, int C1,
                 const float* __restrict__ w, const float* __restrict__ bias,
                 float* __restrict__ out, int KH, int KW, int ph, int pw, int K) {
  __shared__ alignas(16) unsigned short As[64 * ASTRIDE];
  __shared__ alignas(16) unsigned short Bs[32 * BSTRIDE];
  const int tid = threadIdx.x;
  const int mBlock = blockIdx.y * 64;
  const int Cin     = 256 + C1;
  const int ciShift = (C1 > 0) ? 9 : 8;
  const int ciMask  = Cin - 1;
  const int ar  = tid >> 2, ac = (tid & 3) * 8;
  const int bkr = tid >> 3, bng = (tid & 7) * 32;
  const int wave = tid >> 5, lane = tid & 31;
  const int wm = (wave >> 2) * 32, wn = (wave & 3) * 64;

  const int m   = mBlock + ar;
  const int bb  = m >> 12;          // /(64*64)
  const int rem = m & 4095;
  const int hh  = rem >> 6;
  const int wx  = rem & 63;

  v8f acc[2][4];
#pragma unroll
  for (int i = 0; i < 2; i++)
#pragma unroll
    for (int j = 0; j < 4; j++)
#pragma unroll
      for (int v = 0; v < 8; v++) acc[i][j][v] = 0.f;

  for (int kb = 0; kb < K; kb += 32) {
#pragma unroll
    for (int j = 0; j < 8; j++) {
      int kk  = kb + ac + j;
      int tap = kk >> ciShift;
      int ci  = kk & ciMask;
      int ky  = tap / KW;
      int kx  = tap - ky * KW;
      int hy  = hh + ky - ph;
      int wy  = wx + kx - pw;
      float v = 0.f;
      if (hy >= 0 && hy < 64 && wy >= 0 && wy < 64) {
        int pix = (bb << 12) + (hy << 6) + wy;
        v = (ci < 256) ? x0[(size_t)pix * 256 + ci]
                       : x1[(size_t)pix * 256 + (ci - 256)];
      }
      As[ar * ASTRIDE + ac + j] = f2bf(v);
    }
    {
      int kk  = kb + bkr;
      int tap = kk >> ciShift;
      int ci  = kk & ciMask;
      int ky  = tap / KW;
      int kx  = tap - ky * KW;
#pragma unroll
      for (int j = 0; j < 32; j++) {
        int o = bng + j;      // Cout == 256 == BN, nBlock == 0
        Bs[bkr * BSTRIDE + bng + j] = f2bf(w[((size_t)(o * Cin + ci) * KH + ky) * KW + kx]);
      }
    }
    __syncthreads();
    v16bf a0 = frag_a(As, wm, lane), a1 = frag_a(As, wm + 16, lane);
    v16bf bf[4];
#pragma unroll
    for (int f = 0; f < 4; f++) bf[f] = frag_b(Bs, wn + f * 16, lane);
#pragma unroll
    for (int f = 0; f < 4; f++) {
      acc[0][f] = wmma_bf16(a0, bf[f], acc[0][f]);
      acc[1][f] = wmma_bf16(a1, bf[f], acc[1][f]);
    }
    __syncthreads();
  }

#pragma unroll
  for (int fm = 0; fm < 2; fm++)
#pragma unroll
    for (int fn = 0; fn < 4; fn++)
#pragma unroll
      for (int v = 0; v < 8; v++) {
        int mrow = mBlock + wm + fm * 16 + v + ((lane >> 4) << 3);
        int col  = wn + fn * 16 + (lane & 15);
        float x = acc[fm][fn][v] + bias[col];
        out[(size_t)mrow * 256 + col] = activate<ACT>(x);
      }
}

// ---------------- specialized  A_raw = exp(scale * q q^T) ----------------
// q is bf16: A tile staged via global_load_async_to_lds_b128 (ASYNCcnt path),
// B tile via vectorized row reads + transposed ds_store scatter.
__global__ __launch_bounds__(256)
void attn_exp_kernel(const unsigned short* __restrict__ q,
                     const float* __restrict__ scaleArr,
                     float* __restrict__ Aout) {
  __shared__ alignas(16) unsigned short As[64 * ASTRIDE];
  __shared__ alignas(16) unsigned short Bs[32 * BSTRIDE];
  const int tid = threadIdx.x;
  const int z = blockIdx.z;
  const unsigned short* qb = q + (size_t)z * LSEQ * CC;
  float* Ab = Aout + (size_t)z * LSEQ * LSEQ;
  const int mBlock = blockIdx.y * 64;
  const int nBlock = blockIdx.x * 256;
  const int ar  = tid >> 2, aseg = (tid & 3) * 8;   // async A: row, 16-byte segment
  const int wave = tid >> 5, lane = tid & 31;
  const int wm = (wave >> 2) * 32, wn = (wave & 3) * 64;

  const unsigned int ldsA = (unsigned int)(uintptr_t)(As + ar * ASTRIDE + aseg);
  const unsigned short* gA = qb + (size_t)(mBlock + ar) * CC + aseg;
  const unsigned short* gB = qb + (size_t)(nBlock + tid) * CC;

  v8f acc[2][4];
#pragma unroll
  for (int i = 0; i < 2; i++)
#pragma unroll
    for (int j = 0; j < 4; j++)
#pragma unroll
      for (int v = 0; v < 8; v++) acc[i][j][v] = 0.f;

  for (int kb = 0; kb < CC; kb += 32) {
    // A tile: 64 rows x 64B, one async b128 per thread, LDS written by DMA path
    asm volatile("global_load_async_to_lds_b128 %0, %1, off"
                 :: "v"(ldsA), "v"(gA + kb) : "memory");
    // B tile (transposed): thread reads 32 contiguous bf16 of one q row, scatters columns
    union { uint4 q4[4]; unsigned short us[32]; } u;
#pragma unroll
    for (int s = 0; s < 4; s++) u.q4[s] = *(const uint4*)(gB + kb + s * 8);
#pragma unroll
    for (int k = 0; k < 32; k++) Bs[k * BSTRIDE + tid] = u.us[k];
    asm volatile("s_wait_asynccnt 0" ::: "memory");
    __syncthreads();

    v16bf a0 = frag_a(As, wm, lane), a1 = frag_a(As, wm + 16, lane);
    v16bf bf[4];
#pragma unroll
    for (int f = 0; f < 4; f++) bf[f] = frag_b(Bs, wn + f * 16, lane);
#pragma unroll
    for (int f = 0; f < 4; f++) {
      acc[0][f] = wmma_bf16(a0, bf[f], acc[0][f]);
      acc[1][f] = wmma_bf16(a1, bf[f], acc[1][f]);
    }
    __syncthreads();
  }
  const float sc = scaleArr[z];
#pragma unroll
  for (int fm = 0; fm < 2; fm++)
#pragma unroll
    for (int fn = 0; fn < 4; fn++)
#pragma unroll
      for (int v = 0; v < 8; v++) {
        int mrow = mBlock + wm + fm * 16 + v + ((lane >> 4) << 3);
        int col  = nBlock + wn + fn * 16 + (lane & 15);
        Ab[(size_t)mrow * LSEQ + col] = expf(sc * acc[fm][fn][v]);
      }
}

// ---------------- small kernels ----------------
__global__ void zero_kernel(float* p, int n) {
  int i = blockIdx.x * blockDim.x + threadIdx.x;
  if (i < n) p[i] = 0.f;
}

__global__ void ln_kernel(const float* __restrict__ x, const float* __restrict__ g,
                          const float* __restrict__ b, float* __restrict__ y, int rows) {
  int wv = threadIdx.x >> 5, lane = threadIdx.x & 31;
  int row = blockIdx.x * 8 + wv;
  if (row >= rows) return;
  const float* xr = x + (size_t)row * 256;
  float v[8], s = 0.f;
#pragma unroll
  for (int j = 0; j < 8; j++) { v[j] = xr[j * 32 + lane]; s += v[j]; }
  s = wred_sum(s);
  float mean = s * (1.f / 256.f);
  float s2 = 0.f;
#pragma unroll
  for (int j = 0; j < 8; j++) { float d = v[j] - mean; s2 += d * d; }
  s2 = wred_sum(s2);
  float inv = rsqrtf(s2 * (1.f / 256.f) + 1e-5f);
  float* yr = y + (size_t)row * 256;
#pragma unroll
  for (int j = 0; j < 8; j++) {
    int c = j * 32 + lane;
    yr[c] = (v[j] - mean) * inv * g[c] + b[c];
  }
}

__global__ void rowstat_kernel(const float* __restrict__ x, float* __restrict__ normfac,
                               float* __restrict__ meanacc) {
  int wv = threadIdx.x >> 5, lane = threadIdx.x & 31;
  int row = blockIdx.x * 8 + wv;
  const float* xr = x + (size_t)row * 256;
  float s1 = 0.f, s2 = 0.f;
#pragma unroll
  for (int j = 0; j < 8; j++) { float v = xr[j * 32 + lane]; s1 += v; s2 += v * v; }
  s1 = wred_sum(s1);
  s2 = wred_sum(s2);
  if (lane == 0) {
    normfac[row] = sqrtf(s2);
    atomicAdd(&meanacc[row >> 12], s1);
  }
}

__global__ void scale_kernel(const float* meanacc, float* scale) {
  int i = threadIdx.x;
  if (i < 2) {
    float m = meanacc[i] * (1.f / (4096.f * 256.f));
    scale[i] = 9.42477796076938f / (1.f + expf(-m)) + 1.f;   // 3*pi*sigmoid(m)+1
  }
}

__global__ void qpack_kernel(const float* __restrict__ attp, const float* __restrict__ nf,
                             unsigned short* __restrict__ q) {
  int i = blockIdx.x * 256 + threadIdx.x;
  q[i] = f2bf(attp[i] / nf[i >> 8]);
}

__global__ void rowsumA_kernel(const float* __restrict__ A, float* __restrict__ rowsum,
                               int* __restrict__ maxbits) {
  __shared__ float ss[8], sm[8];
  int row = blockIdx.x;
  size_t base = (size_t)row * 4096;
  float s = 0.f, m = -1e30f;
#pragma unroll
  for (int j = 0; j < 16; j++) {
    float v = A[base + threadIdx.x + j * 256];
    s += v; m = fmaxf(m, v);
  }
  s = wred_sum(s); m = wred_max(m);
  int wv = threadIdx.x >> 5, lane = threadIdx.x & 31;
  if (lane == 0) { ss[wv] = s; sm[wv] = m; }
  __syncthreads();
  if (threadIdx.x == 0) {
    float S = 0.f, M = -1e30f;
    for (int i = 0; i < 8; i++) { S += ss[i]; M = fmaxf(M, sm[i]); }
    rowsum[row] = S;
    atomicMax(maxbits, __float_as_int(M));   // all values > 0
  }
}

__global__ void dvec_kernel(const float* rowsum, const int* maxbits, float* d) {
  int i = blockIdx.x * 256 + threadIdx.x;
  float invMax = 1.f / __int_as_float(*maxbits);
  d[i] = 1.f / (sqrtf(rowsum[i] * invMax) + 1e-6f);
}

__global__ void normA_kernel(float* __restrict__ A, const float* __restrict__ d,
                             const int* __restrict__ maxbits) {
  int row = blockIdx.x;
  float invMax = 1.f / __int_as_float(*maxbits);
  float fac = invMax * d[row];
  const float* dcol = d + ((row >> 12) << 12);
  size_t base = (size_t)row * 4096;
#pragma unroll
  for (int j = 0; j < 16; j++) {
    int c = threadIdx.x + j * 256;
    A[base + c] *= fac * dcol[c];
  }
}

__global__ void copy_kernel(float* dst, const float* src, int n) {
  int i = blockIdx.x * 256 + threadIdx.x;
  if (i < n) dst[i] = src[i];
}
__global__ void mul_kernel(float* out, const float* a, const float* b, int n) {
  int i = blockIdx.x * 256 + threadIdx.x;
  if (i < n) out[i] = a[i] * b[i];
}
__global__ void gruupd_kernel(float* h, const float* z, const float* q, int n) {
  int i = blockIdx.x * 256 + threadIdx.x;
  if (i < n) h[i] = (1.f - z[i]) * h[i] + z[i] * q[i];
}

// ---------------- host ----------------
extern "C" void kernel_launch(void* const* d_in, const int* in_sizes, int n_in,
                              void* d_out, int out_size, void* d_ws, size_t ws_size,
                              hipStream_t stream) {
  (void)in_sizes; (void)n_in; (void)out_size; (void)ws_size;
  const float* att     = (const float*)d_in[0];
  const float* value   = (const float*)d_in[1];
  const float* att_w1  = (const float*)d_in[2];
  const float* att_w2  = (const float*)d_in[3];
  const float* v_w     = (const float*)d_in[4];
  const float* merge_w = (const float*)d_in[5];
  const float* n1_g    = (const float*)d_in[6];
  const float* n1_b    = (const float*)d_in[7];
  const float* mlp_w1  = (const float*)d_in[8];
  const float* mlp_w2  = (const float*)d_in[9];
  const float* mlp_w3  = (const float*)d_in[10];
  const float* n2_g    = (const float*)d_in[11];
  const float* n2_b    = (const float*)d_in[12];
  const float* au_proj = (const float*)d_in[13];
  const float* au_ng   = (const float*)d_in[14];
  const float* au_nb   = (const float*)d_in[15];
  const float* au_c1w  = (const float*)d_in[16];
  const float* au_c1b  = (const float*)d_in[17];
  const float* au_c2w  = (const float*)d_in[18];
  const float* au_c2b  = (const float*)d_in[19];
  const float* au_c3w  = (const float*)d_in[20];
  const float* au_c3b  = (const float*)d_in[21];
  // GRU weights: base 22 (au_), 34 (g_): z1w z1b r1w r1b q1w q1b z2w z2b r2w r2b q2w q2b

  float* out    = (float*)d_out;
  float* outVal = out;
  float* outAtt = out + (size_t)NB * LSEQ * CC;
  float* outA   = out + (size_t)2 * NB * LSEQ * CC;

  char* W = (char*)d_ws;
  size_t off = 0;
  auto take = [&](size_t bytes) -> char* {
    char* r = W + off;
    off = (off + bytes + 255) & ~(size_t)255;
    return r;
  };
  const size_t SZ = (size_t)MTOK * CC * sizeof(float);   // 8 MB per token buffer
  float* val_proj = (float*)take(SZ);
  float* attp     = (float*)take(SZ);
  float* tbuf     = (float*)take(SZ);
  float* msg1     = (float*)take(SZ);
  float* msg2     = (float*)take(SZ);
  float* msgp     = (float*)take(SZ);
  float* bufA     = (float*)take(SZ);
  float* bufB     = (float*)take(SZ);
  float* zg       = (float*)take(SZ);
  float* rg       = (float*)take(SZ);
  float* qc       = (float*)take(SZ);
  float* rhb      = (float*)take(SZ);
  float* hatt     = (float*)take(SZ);
  float* hval     = (float*)take(SZ);
  unsigned short* qbf = (unsigned short*)take((size_t)MTOK * CC * 2);
  float* normfac  = (float*)take(MTOK * 4);
  float* rowsum   = (float*)take(MTOK * 4);
  float* dv       = (float*)take(MTOK * 4);
  float* meanacc  = (float*)take(256);   // [0..1]=mean acc, [2..3]=scale, [4]=maxbits
  float* scalev   = meanacc + 2;
  int*   maxbits  = (int*)(meanacc + 4);
  float* h1       = (float*)take((size_t)2048 * 2048 * 4);
  float* h2       = (float*)take((size_t)2048 * 2048 * 4);

  const int NE = MTOK * CC;            // 2,097,152 elementwise size
  dim3 blk(256);

  zero_kernel<<<1, 32, 0, stream>>>(meanacc, 8);

  // projections
  gemm_kernel<0><<<dim3(1,128,1), blk, 0, stream>>>(value, value, CC, v_w, nullptr, val_proj,
                                                    MTOK, CC, CC, 0, 0, 0);
  gemm_kernel<1><<<dim3(1,128,1), blk, 0, stream>>>(att, att, CC, att_w1, nullptr, tbuf,
                                                    MTOK, CC, CC, 0, 0, 0);
  gemm_kernel<0><<<dim3(1,128,1), blk, 0, stream>>>(tbuf, tbuf, CC, att_w2, nullptr, attp,
                                                    MTOK, CC, CC, 0, 0, 0);
  // norm_fac + batch mean -> scale -> q (bf16)
  rowstat_kernel<<<1024, blk, 0, stream>>>(attp, normfac, meanacc);
  scale_kernel<<<1, 32, 0, stream>>>(meanacc, scalev);
  qpack_kernel<<<NE / 256, blk, 0, stream>>>(attp, normfac, qbf);

  // A_raw = exp(scale * q q^T) directly into output A slot, then normalize in-place
  attn_exp_kernel<<<dim3(16,64,2), blk, 0, stream>>>(qbf, scalev, outA);
  rowsumA_kernel<<<MTOK, blk, 0, stream>>>(outA, rowsum, maxbits);
  dvec_kernel<<<32, blk, 0, stream>>>(rowsum, maxbits, dv);
  normA_kernel<<<MTOK, blk, 0, stream>>>(outA, dv, maxbits);

  // message = LN1( (A @ val_proj) @ merge_w )
  gemm_kernel<0><<<dim3(1,64,2), blk, 0, stream>>>(outA, outA, LSEQ, val_proj, nullptr, tbuf,
                                                   LSEQ, CC, LSEQ,
                                                   (long long)LSEQ * LSEQ,
                                                   (long long)LSEQ * CC,
                                                   (long long)LSEQ * CC);
  gemm_kernel<0><<<dim3(1,128,1), blk, 0, stream>>>(tbuf, tbuf, CC, merge_w, nullptr, bufA,
                                                    MTOK, CC, CC, 0, 0, 0);
  ln_kernel<<<1024, blk, 0, stream>>>(bufA, n1_g, n1_b, msg1, MTOK);

  // FFN on concat([value, msg1]) in 4 row-chunks of 2048
  for (int c = 0; c < 4; c++) {
    const size_t ro = (size_t)c * 2048 * CC;
    gemm_kernel<2><<<dim3(8,32,1), blk, 0, stream>>>(value + ro, msg1 + ro, CC, mlp_w1, nullptr,
                                                     h1, 2048, 2048, 512, 0, 0, 0);
    gemm_kernel<2><<<dim3(8,32,1), blk, 0, stream>>>(h1, h1, 2048, mlp_w2, nullptr,
                                                     h2, 2048, 2048, 2048, 0, 0, 0);
    gemm_kernel<0><<<dim3(1,32,1), blk, 0, stream>>>(h2, h2, 2048, mlp_w3, nullptr,
                                                     tbuf + ro, 2048, CC, 2048, 0, 0, 0);
  }
  ln_kernel<<<1024, blk, 0, stream>>>(tbuf, n2_g, n2_b, msg2, MTOK);

  // AttUp: proj + LN, conv1x1 + 2x conv3x3 (all gelu)
  gemm_kernel<0><<<dim3(1,128,1), blk, 0, stream>>>(msg2, msg2, CC, au_proj, nullptr, bufA,
                                                    MTOK, CC, CC, 0, 0, 0);
  ln_kernel<<<1024, blk, 0, stream>>>(bufA, au_ng, au_nb, msgp, MTOK);
  conv_kernel<2><<<dim3(1,128,1), blk, 0, stream>>>(att,  msgp, 256, au_c1w, au_c1b, bufA,
                                                    1, 1, 0, 0, 512);
  conv_kernel<2><<<dim3(1,128,1), blk, 0, stream>>>(bufA, bufA, 0,   au_c2w, au_c2b, bufB,
                                                    3, 3, 1, 1, 2304);
  conv_kernel<2><<<dim3(1,128,1), blk, 0, stream>>>(bufB, bufB, 0,   au_c3w, au_c3b, bufA,
                                                    3, 3, 1, 1, 2304);

  // SepConvGRU: h updated in-place; x = conv-stack output (att) / msg2 (value)
  auto gru = [&](float* h, const float* x, int wbase) {
    const float* w[12];
    for (int i = 0; i < 12; i++) w[i] = (const float*)d_in[wbase + i];
    // horizontal (1,5), pad (0,2)
    conv_kernel<3><<<dim3(1,128,1), blk, 0, stream>>>(h,   x, 256, w[0], w[1], zg, 1,5,0,2, 2560);
    conv_kernel<3><<<dim3(1,128,1), blk, 0, stream>>>(h,   x, 256, w[2], w[3], rg, 1,5,0,2, 2560);
    mul_kernel<<<NE/256, blk, 0, stream>>>(rhb, rg, h, NE);
    conv_kernel<4><<<dim3(1,128,1), blk, 0, stream>>>(rhb, x, 256, w[4], w[5], qc, 1,5,0,2, 2560);
    gruupd_kernel<<<NE/256, blk, 0, stream>>>(h, zg, qc, NE);
    // vertical (5,1), pad (2,0)
    conv_kernel<3><<<dim3(1,128,1), blk, 0, stream>>>(h,   x, 256, w[6], w[7], zg, 5,1,2,0, 2560);
    conv_kernel<3><<<dim3(1,128,1), blk, 0, stream>>>(h,   x, 256, w[8], w[9], rg, 5,1,2,0, 2560);
    mul_kernel<<<NE/256, blk, 0, stream>>>(rhb, rg, h, NE);
    conv_kernel<4><<<dim3(1,128,1), blk, 0, stream>>>(rhb, x, 256, w[10], w[11], qc, 5,1,2,0, 2560);
    gruupd_kernel<<<NE/256, blk, 0, stream>>>(h, zg, qc, NE);
  };

  copy_kernel<<<NE/256, blk, 0, stream>>>(hatt, att, NE);
  gru(hatt, bufA, 22);
  copy_kernel<<<NE/256, blk, 0, stream>>>(outAtt, hatt, NE);

  copy_kernel<<<NE/256, blk, 0, stream>>>(hval, value, NE);
  gru(hval, msg2, 34);
  copy_kernel<<<NE/256, blk, 0, stream>>>(outVal, hval, NE);
}
